// PositionCloser_20985210208956
// MI455X (gfx1250) — 8-bit matrix operand order repaired and checked by reading, compiled, not run on hardware
//
#include <hip/hip_runtime.h>
#include <math.h>

#define Bsz 64
#define Wsz 32
#define NSEQ (Bsz*Wsz)
#define Hh 256
#define OHh 512
#define NG 1024
#define CHUNK 64
#define NCHUNKS 8
#define NSTEPS (CHUNK*NCHUNKS)
#define MTILE 16
#define NWG (NSEQ/MTILE)
#define THREADS 512

typedef __attribute__((ext_vector_type(16))) int    v16i_t;
typedef __attribute__((ext_vector_type(8)))  float  v8f_t;

__device__ __forceinline__ unsigned char f2fp8(float f){
  unsigned int u = __builtin_bit_cast(unsigned int, f);
  unsigned int s = (u >> 24) & 0x80u;
  unsigned int a = u & 0x7fffffffu;
  if (a > 0x43E00000u) return (unsigned char)(s | 0x7Eu);
  int e = (int)(a >> 23) - 127;
  if (e >= -6){
    unsigned int v  = ((unsigned int)(e + 7) << 3) | ((a >> 20) & 7u);
    unsigned int g  = (a >> 19) & 1u;
    unsigned int st = (a & 0x7FFFFu) ? 1u : 0u;
    v += g & (st | (v & 1u));
    if (v > 0x7Eu) v = 0x7Eu;
    return (unsigned char)(s | v);
  }
  float m = __builtin_bit_cast(float, a) * 512.0f;
  int d = (int)(m + 0.5f);
  return (unsigned char)(s | (unsigned int)d);
}
__device__ __forceinline__ float sigf(float x){ return 1.0f/(1.0f + __expf(-x)); }

__global__ void conv_whh_fp8(const float* __restrict__ W_hh,
                             unsigned char* __restrict__ W8, int n){
  int i = blockIdx.x * blockDim.x + threadIdx.x;
  if (i < n) W8[i] = f2fp8(W_hh[i]);
}

__global__ __launch_bounds__(THREADS, 1)
void lstm_main(const int*   __restrict__ inds,
               const float* __restrict__ input,
               const float* __restrict__ p,
               const float* __restrict__ ls_probs,
               const float* __restrict__ open_probs,
               const int*   __restrict__ open_slices,
               const float* __restrict__ open_output,
               const float* __restrict__ W_ih,
               const float* __restrict__ b_ih,
               const float* __restrict__ b_hh,
               const float* __restrict__ W_prob,
               const float* __restrict__ b_prob,
               const float* __restrict__ W_o2c,
               const float* __restrict__ b_o2c,
               const unsigned char* __restrict__ W8,
               float* __restrict__ loss_part,
               float* __restrict__ cum_tot,
               float* __restrict__ out_last,
               float* __restrict__ coef_last)
{
  __shared__ unsigned char hA8[MTILE*Hh];
  __shared__ float hF[MTILE*Hh];
  __shared__ float cS[MTILE*Hh];
  __shared__ float gates[MTILE*NG];
  __shared__ float wih0[NG], wih1[NG], bsum[NG];
  __shared__ float wpr[Hh];
  __shared__ float xs0[MTILE], xs1[MTILE];
  __shared__ float pp[MTILE][32];
  __shared__ float st_nc[MTILE], st_cum[MTILE], st_loss[MTILE];
  __shared__ float st_ob[MTILE], st_os[MTILE], st_lsp[MTILE], st_opp[MTILE];
  __shared__ int   st_base[MTILE];

  const int tid  = threadIdx.x;
  const int lane = tid & 31;
  const int wave = tid >> 5;
  const int seq0 = blockIdx.x * MTILE;

  for (int i = tid; i < NG; i += THREADS){
    wih0[i] = W_ih[i*2+0];
    wih1[i] = W_ih[i*2+1];
    bsum[i] = b_ih[i] + b_hh[i];
  }
  if (tid < Hh) wpr[tid] = W_prob[tid];

  for (int e = 0; e < 8; ++e){
    int idx = tid + e*THREADS;
    int m = idx >> 8, n = idx & 255;
    const float* oo = open_output + (size_t)(seq0 + m) * OHh;
    const float* wr = W_o2c      + (size_t)n * OHh;
    float acc = b_o2c[n];
    for (int k = 0; k < OHh; ++k) acc += oo[k] * wr[k];
    hF[m*Hh + n]  = acc;
    hA8[m*Hh + n] = f2fp8(acc);
    cS[m*Hh + n]  = 0.f;
  }

  if (tid < MTILE){
    int s = seq0 + tid;
    int b = s / Wsz, w = s % Wsz;
    int osl = open_slices[s];
    st_ob[tid]  = __logf(p[osl*2+0]);
    st_os[tid]  = __logf(p[osl*2+1]);
    st_lsp[tid] = ls_probs[s];
    st_opp[tid] = open_probs[s];
    st_base[tid]= inds[b] + w;
    st_nc[tid]  = 1.f;
    st_cum[tid] = 0.f;
    st_loss[tid]= 0.f;
  }

  const int nb0 = wave * 64;
  const unsigned int* W8u = (const unsigned int*)W8;
  v16i_t Bf[4][2];
  #pragma unroll
  for (int nt = 0; nt < 4; ++nt){
    int n = nb0 + nt*16 + (lane & 15);
    #pragma unroll
    for (int kt = 0; kt < 2; ++kt){
      int base = n*Hh + kt*128 + 16*(lane >> 4);
      #pragma unroll
      for (int j = 0; j < 16; ++j){
        int off = 32*(j >> 2) + 4*(j & 3);
        Bf[nt][kt][j] = (int)W8u[(base + off) >> 2];
      }
    }
  }

  __syncthreads();

  for (int t = 0; t < NSTEPS; ++t){
    if (tid < MTILE){
      int bi  = st_base[tid];
      int ii  = bi + t;
      int ii0 = bi + (t & ~(CHUNK-1));
      xs0[tid] = input[ii*2+0] - input[ii0*2+0];
      xs1[tid] = input[ii*2+1];
    }

    {
      v8f_t acc[4];
      #pragma unroll
      for (int nt = 0; nt < 4; ++nt) acc[nt] = (v8f_t){0,0,0,0,0,0,0,0};
      const int m  = lane & 15;
      const int kh = 8 * (lane >> 4);
      const unsigned int* hrow = (const unsigned int*)(hA8 + m*Hh);
      #pragma unroll
      for (int kt = 0; kt < 2; ++kt){
        v16i_t a;
        #pragma unroll
        for (int j = 0; j < 16; ++j){
          int off = 128*kt + 32*(j >> 2) + 4*(j & 3) + 2*kh;
          a[j] = (int)hrow[off >> 2];
        }
#if defined(__gfx1250__)
        #pragma unroll
        for (int nt = 0; nt < 4; ++nt){
          acc[nt] = __builtin_amdgcn_wmma_f32_16x16x128_fp8_fp8(
                      a, Bf[nt][kt], (short)0, acc[nt], false, false);
        }
#else
        (void)a;
#endif
      }
      const int mrow = 8 * (lane >> 4);
      const int ncol = lane & 15;
      #pragma unroll
      for (int nt = 0; nt < 4; ++nt){
        #pragma unroll
        for (int v = 0; v < 8; ++v){
          gates[(mrow+v)*NG + nb0 + nt*16 + ncol] = acc[nt][v];
        }
      }
    }
    __syncthreads();

    #pragma unroll
    for (int e = 0; e < 8; ++e){
      int idx = tid + e*THREADS;
      int m = idx >> 8, n = idx & 255;
      float x0 = xs0[m], x1 = xs1[m];
      int gi = n, gf = 256 + n, gg = 512 + n, go = 768 + n;
      float gvi = gates[m*NG+gi] + x0*wih0[gi] + x1*wih1[gi] + bsum[gi];
      float gvf = gates[m*NG+gf] + x0*wih0[gf] + x1*wih1[gf] + bsum[gf];
      float gvg = gates[m*NG+gg] + x0*wih0[gg] + x1*wih1[gg] + bsum[gg];
      float gvo = gates[m*NG+go] + x0*wih0[go] + x1*wih1[go] + bsum[go];
      float c = sigf(gvf) * cS[m*Hh+n] + sigf(gvi) * tanhf(gvg);
      float h = sigf(gvo) * tanhf(c);
      cS[m*Hh+n]  = c;
      hF[m*Hh+n]  = h;
      hA8[m*Hh+n] = f2fp8(h);
    }
    __syncthreads();

    {
      int m = tid >> 5, j = tid & 31;
      float s = 0.f;
      for (int k = j; k < Hh; k += 32) s += hF[m*Hh+k] * wpr[k];
      pp[m][j] = s;
    }
    __syncthreads();

    if (tid < MTILE){
      float d = b_prob[0];
      #pragma unroll
      for (int j = 0; j < 32; ++j) d += pp[tid][j];
      float pr  = sigf(d);
      float nc  = st_nc[tid];
      float out = pr * nc;
      st_nc[tid]  = nc * (1.f - pr);
      float cum = st_cum[tid] + out;
      st_cum[tid] = cum;
      int ii = st_base[tid] + t;
      float cs = __logf(p[ii*2+1]);
      float coef = st_opp[tid] *
                   (st_lsp[tid]*(cs - st_ob[tid]) + (1.f - st_lsp[tid])*(st_os[tid] - cs));
      if (t < NSTEPS-1){
        st_loss[tid] += coef * out;
      } else {
        int s = seq0 + tid;
        loss_part[s] = st_loss[tid];
        cum_tot[s]   = cum;
        out_last[s]  = out;
        coef_last[s] = coef;
      }
    }
  }
}

__global__ __launch_bounds__(1024)
void finalize(const float* __restrict__ loss_part,
              const float* __restrict__ cum_tot,
              const float* __restrict__ out_last,
              const float* __restrict__ coef_last,
              float* __restrict__ d_out)
{
  __shared__ float red[1024];
  __shared__ int   satred[1024];
  int tid = threadIdx.x;
  const float tol = 1e-8f + 1e-5f;

  int sat = 1; float l = 0.f;
  for (int s = tid; s < NSEQ; s += 1024){
    sat &= (fabsf(cum_tot[s] - 1.f) <= tol) ? 1 : 0;
    l   += loss_part[s];
  }
  satred[tid] = sat; red[tid] = l;
  __syncthreads();
  for (int off = 512; off > 0; off >>= 1){
    if (tid < off){ satred[tid] &= satred[tid+off]; red[tid] += red[tid+off]; }
    __syncthreads();
  }
  float base = red[0];
  int satAll = satred[0];
  __syncthreads();

  float l2 = 0.f;
  for (int s = tid; s < NSEQ; s += 1024){
    float pl = satAll ? out_last[s] : (1.f - cum_tot[s]);
    l2 += coef_last[s] * pl;
  }
  red[tid] = l2;
  __syncthreads();
  for (int off = 512; off > 0; off >>= 1){
    if (tid < off) red[tid] += red[tid+off];
    __syncthreads();
  }
  if (tid == 0) d_out[0] = -(base + red[0]);
}

extern "C" void kernel_launch(void* const* d_in, const int* in_sizes, int n_in,
                              void* d_out, int out_size, void* d_ws, size_t ws_size,
                              hipStream_t stream)
{
  const int*   inds   = (const int*)  d_in[0];
  const float* input  = (const float*)d_in[1];
  const float* p      = (const float*)d_in[2];
  const float* lsp    = (const float*)d_in[3];
  const float* opp    = (const float*)d_in[4];
  const int*   osl    = (const int*)  d_in[5];
  const float* oout   = (const float*)d_in[6];
  const float* W_ih   = (const float*)d_in[7];
  const float* W_hh   = (const float*)d_in[8];
  const float* b_ih   = (const float*)d_in[9];
  const float* b_hh   = (const float*)d_in[10];
  const float* W_prob = (const float*)d_in[11];
  const float* b_prob = (const float*)d_in[12];
  const float* W_o2c  = (const float*)d_in[13];
  const float* b_o2c  = (const float*)d_in[14];

  char* ws = (char*)d_ws;
  unsigned char* W8 = (unsigned char*)ws;
  float* loss_part = (float*)(ws + 256*1024);
  float* cum_tot   = loss_part + NSEQ;
  float* out_last  = cum_tot   + NSEQ;
  float* coef_last = out_last  + NSEQ;

  conv_whh_fp8<<<(NG*Hh + 255)/256, 256, 0, stream>>>(W_hh, W8, NG*Hh);

  lstm_main<<<NWG, THREADS, 0, stream>>>(inds, input, p, lsp, opp, osl, oout,
      W_ih, b_ih, b_hh, W_prob, b_prob, W_o2c, b_o2c, W8,
      loss_part, cum_tot, out_last, coef_last);

  finalize<<<1, 1024, 0, stream>>>(loss_part, cum_tot, out_last, coef_last,
                                   (float*)d_out);
}
